// NonlinearFilter_68693706932920
// MI455X (gfx1250) — matrix operand order repaired and checked by reading, not compiled, not run on hardware
//
#include <hip/hip_runtime.h>
#include <math.h>

typedef __attribute__((ext_vector_type(16))) _Float16 v16h;
typedef __attribute__((ext_vector_type(8)))  float    v8f;
typedef __attribute__((ext_vector_type(2)))  float    v2f;

#define NB 64
#define NT 128
#define NL 128
#define NR 16
#define NS 32
#define NH 256
#define EPSF 1e-8f

__device__ __forceinline__ unsigned hashu(unsigned x) {
  x ^= x >> 16; x *= 0x7feb352dU;
  x ^= x >> 15; x *= 0x846ca68bU;
  x ^= x >> 16; return x;
}

// deterministic standard normal, keyed like the reference's draw shapes
__device__ __forceinline__ float rngn(unsigned role, unsigned t, unsigned s,
                                      unsigned b, unsigned i) {
  unsigned k1 = hashu(role * 0x9E3779B9u ^ (t * 0x85EBCA6Bu) ^ 0x5bd1e995u);
  k1 = hashu(k1 ^ (s * 0xC2B2AE35u) ^ (b * 0x27D4EB2Fu) ^ (i * 0x165667B1u));
  unsigned k2 = hashu(k1 + 0x68E31DA4u);
  float u1 = ((float)(k1 >> 8) + 1.0f) * (1.0f / 16777216.0f); // (0,1]
  float u2 = (float)(k2 >> 8) * (1.0f / 16777216.0f);
  return sqrtf(-2.0f * logf(u1)) * cosf(6.28318530718f * u2);
}

__device__ __forceinline__ float softplusf(float x) {
  return (x > 20.0f) ? x : log1pf(expf(x));
}

// cooperative in-place lower Cholesky of n x n (ld = row stride), 256 threads
__device__ __forceinline__ void chol_lower(float* A, int n, int ld, int tid) {
  const int wv = tid >> 5, lane = tid & 31;
  for (int kk = 0; kk < n; ++kk) {
    __syncthreads();
    if (tid == 0) A[kk * ld + kk] = sqrtf(fmaxf(A[kk * ld + kk], 1e-20f));
    __syncthreads();
    if (tid > kk && tid < n) A[tid * ld + kk] /= A[kk * ld + kk];
    __syncthreads();
    for (int i = kk + 1 + wv; i < n; i += 8)
      for (int j = kk + 1 + lane; j <= i; j += 32)
        A[i * ld + j] -= A[i * ld + kk] * A[j * ld + kk];
  }
  __syncthreads();
}

// ---------------------------------------------------------------------------
// Pre-swizzle W1/W2 (f32 row-major) into f16 WMMA B-fragment layout in d_ws.
// B tile (32K x 16N): lanes 0-15 hold K=0..7,16..23, lanes 16-31 hold
// K=8..15,24..31 (half index = K%8 + 8*(K>=16)); N = lane%16. One v16h (32B) per lane.
// ---------------------------------------------------------------------------
__global__ __launch_bounds__(256) void nlf_swizzle(const float* __restrict__ W1,
                                                   const float* __restrict__ W2,
                                                   _Float16* __restrict__ ws) {
  int gid = blockIdx.x * 256 + threadIdx.x;
  for (int idx = gid; idx < NL * NH; idx += NB * 256) {   // W1: K-dim=NL rows
    int kk = idx / NH, nn = idx % NH;
    int frag = (kk >> 5) * (NH / 16) + (nn >> 4);
    int lane = (nn & 15) + (((kk & 15) >= 8) ? 16 : 0);
    int h = (kk & 7) + ((kk & 16) ? 8 : 0);
    ws[(frag * 32 + lane) * 16 + h] = (_Float16)W1[idx];
  }
  _Float16* ws2 = ws + NL * NH;
  for (int idx = gid; idx < NH * NL; idx += NB * 256) {   // W2: K-dim=NH rows
    int kk = idx / NL, nn = idx % NL;
    int frag = (kk >> 5) * (NL / 16) + (nn >> 4);
    int lane = (nn & 15) + (((kk & 15) >= 8) ? 16 : 0);
    int h = (kk & 7) + ((kk & 16) ? 8 : 0);
    ws2[(frag * 32 + lane) * 16 + h] = (_Float16)W2[idx];
  }
}

// ---------------------------------------------------------------------------
// Main filter: one block per batch chain b, 256 threads = 8 wave32.
// ---------------------------------------------------------------------------
__global__ __launch_bounds__(256, 1) void nlf_filter(
    const float* __restrict__ kg, const float* __restrict__ Kg,
    const float* __restrict__ log_Q, const float* __restrict__ m0g,
    const float* __restrict__ log_Q0, const float* __restrict__ b1g,
    const float* __restrict__ b2g, const _Float16* __restrict__ wsW,
    float* __restrict__ out) {
  const int b = blockIdx.x;
  const int tid = threadIdx.x;
  const int wv = tid >> 5, lane = tid & 31;
  const int ml = lane & 15;              // M/N index inside f32-WMMA fragments
  const int kh = (lane >> 4) << 1;       // K sub-offset {0,2} inside 4-K step

  // ------------------------- LDS (~128 KB) --------------------------------
  __shared__ float sZ[NS][NL];                 // z state (f32)
  __shared__ __align__(32) _Float16 sH[2 * 8 * 32 * 16]; // hidden A-frags
  __shared__ float sMT[NS][NL];                // m_theta; reused as z_p_c
  __shared__ float sUn[NS][NL];                // unscaled
  __shared__ float sMc[NL][NS + 1];            // M_c (L x S)
  __shared__ float sKt[NL][NR];                // K_t
  __shared__ float sKPsi[NL][NR];              // K @ Psi_f
  __shared__ float sA32[NS][NS + 1];           // I+G / Cp (in place)
  __shared__ float sCf[NR][NR + 1];            // I+KPK / Cf (in place)
  __shared__ float sInv[NR][NR + 1];           // Cf^{-1} (lower)
  __shared__ float sKtM[NR][NS + 1];
  __shared__ float sWp1[NS][NS + 1];
  __shared__ float sWf[NS][NR];
  __shared__ float sV1[NS][NR];
  __shared__ float sY[NS][NR];
  __shared__ float sX[NS][NR];
  __shared__ float sTS[NS][NS + 1];
  __shared__ float sVecA[NL];                  // m_p
  __shared__ float sVecB[NL];                  // h
  __shared__ float sVecC[NL];                  // u1h
  __shared__ float sVecD[NL];                  // m_f
  __shared__ float sVecE[NL];                  // trip
  __shared__ float sQd[NL], sQi[NL], sQs[NL], sPp[NL];
  __shared__ float skt[NL];
  __shared__ float sB1[NH], sB2[NL];
  __shared__ float sSa[NS], sSb[NS];
  __shared__ float sRed[256];
  __shared__ float sScal[8];
  float (*sZp)[NL] = sMT;                      // z_p_c aliases m_theta

  float* z_all = out;
  float* m_all = out + (size_t)NS * NB * NT * NL;
  float* kl_all = m_all + (size_t)NB * NT * NL;

  const v16h* wsW1v = (const v16h*)wsW;
  const v16h* wsW2v = wsW1v + 64 * 32;
  const float sqrtSinv = 1.0f / sqrtf((float)NS);

  auto block_reduce_to = [&](float v, int slot) {
    sRed[tid] = v;
    __syncthreads();
    if (tid == 0) {
      float a_ = 0.f;
      for (int i = 0; i < 256; ++i) a_ += sRed[i];
      sScal[slot] = a_;
    }
    __syncthreads();
  };

  // ----- constants -----
  if (tid < NL) {
    float qd = softplusf(log_Q[tid]);
    sQd[tid] = qd; sQi[tid] = 1.0f / qd; sQs[tid] = sqrtf(qd);
    sPp[tid] = softplusf(log_Q0[tid]);
    sB2[tid] = b2g[tid];
  }
  if (tid < NH) sB1[tid] = b1g[tid];
  for (int idx = tid; idx < NL * NR; idx += 256)
    sKt[idx >> 4][idx & 15] = Kg[(size_t)(b * NT) * NL * NR + idx];
  if (tid < NL) skt[tid] = kg[(size_t)(b * NT) * NL + tid];
  __syncthreads();

  // ======================= t = 0 initialization ===========================
  for (int idx = tid; idx < NS * NL; idx += 256) {   // z_p_c0 = sqrt(Pp)*w_p0
    int s = idx >> 7, l = idx & 127;
    sZp[s][l] = sqrtf(sPp[l]) * rngn(0u, 0u, s, b, l);
  }
  if (tid < NL) sVecB[tid] = m0g[tid] / sPp[tid] + skt[tid];   // h0
  __syncthreads();
  for (int idx = tid; idx < NR * NR; idx += 256) {   // I + K0^T Pp K0
    int r = idx >> 4, q = idx & 15;
    float a_ = (r == q) ? 1.0f : 0.0f;
    for (int l = 0; l < NL; ++l) a_ += sKt[l][r] * sPp[l] * sKt[l][q];
    sCf[r][q] = a_;
  }
  chol_lower(&sCf[0][0], NR, NR + 1, tid);
  if (tid < NR) {                                    // inv = Cf^{-1} (lower)
    int c = tid; float x[NR];
    for (int i = 0; i < NR; ++i) {
      if (i < c) { x[i] = 0.f; continue; }
      float s_ = (i == c) ? 1.f : 0.f;
      for (int j = c; j < i; ++j) s_ -= sCf[i][j] * x[j];
      x[i] = s_ / sCf[i][i];
    }
    for (int i = 0; i < NR; ++i) sInv[i][c] = x[i];
  }
  __syncthreads();
  if (tid < NR) {                                    // a[r] = sum Pp*K0*h
    float a_ = 0.f;
    for (int l = 0; l < NL; ++l) a_ += sPp[l] * sKt[l][tid] * sVecB[l];
    sSa[tid] = a_;
  }
  __syncthreads();
  if (tid < NR) {                                    // y = inv * a
    float a_ = 0.f;
    for (int j = 0; j <= tid; ++j) a_ += sInv[tid][j] * sSa[j];
    sSb[tid] = a_;
  }
  __syncthreads();
  if (tid < NR) {                                    // ps = inv^T * y
    float a_ = 0.f;
    for (int j = tid; j < NR; ++j) a_ += sInv[j][tid] * sSb[j];
    sSa[tid] = a_;
  }
  __syncthreads();
  if (tid < NL) {                                    // m_f0
    float a_ = sVecB[tid];
    for (int r = 0; r < NR; ++r) a_ -= sKt[tid][r] * sSa[r];
    sVecD[tid] = sPp[tid] * a_;
  }
  for (int idx = tid; idx < NS * NR; idx += 256) {   // v1 = K0^T z_p_c + w_f0
    int s = idx >> 4, r = idx & 15;
    float a_ = rngn(1u, 0u, s, b, r);
    for (int l = 0; l < NL; ++l) a_ += sKt[l][r] * sZp[s][l];
    sV1[s][r] = a_;
  }
  __syncthreads();
  for (int idx = tid; idx < NS * NR; idx += 256) {
    int s = idx >> 4, r = idx & 15;
    float a_ = 0.f;
    for (int j = 0; j <= r; ++j) a_ += sInv[r][j] * sV1[s][j];
    sY[s][r] = a_;
  }
  __syncthreads();
  for (int idx = tid; idx < NS * NR; idx += 256) {
    int s = idx >> 4, r = idx & 15;
    float a_ = 0.f;
    for (int j = r; j < NR; ++j) a_ += sInv[j][r] * sY[s][j];
    sX[s][r] = a_;
  }
  for (int idx = tid; idx < NL * NR; idx += 256) {   // K_Psi0
    int l = idx >> 4, r = idx & 15;
    float a_ = 0.f;
    for (int q = 0; q <= r; ++q) a_ += sKt[l][q] * sInv[r][q];
    sKPsi[l][r] = a_;
  }
  __syncthreads();
  for (int idx = tid; idx < NS * NL; idx += 256) {   // z_f0
    int s = idx >> 7, l = idx & 127;
    float a_ = 0.f;
    for (int r = 0; r < NR; ++r) a_ += sKt[l][r] * sX[s][r];
    float zf = sVecD[l] + sZp[s][l] - sPp[l] * a_;
    sZ[s][l] = zf;
    z_all[(((size_t)s * NB + b) * NT + 0) * NL + l] = zf;
  }
  {
    float accT = 0.f;
    for (int idx = tid; idx < NL * NR; idx += 256) {
      int l = idx >> 4, r = idx & 15;
      float kp = sKPsi[l][r];
      accT += sPp[l] * kp * kp;
    }
    block_reduce_to(accT, 0);
    float accq = 0.f;
    if (tid < NL) {
      float dm = sVecD[tid] - m0g[tid];
      accq = dm * dm / sPp[tid];
    }
    block_reduce_to(accq, 1);
    if (tid == 0) {
      float ld = 0.f;
      for (int i = 0; i < NR; ++i) ld += logf(sInv[i][i] + EPSF);
      kl_all[(size_t)b * NT] =
          0.5f * (sScal[1] + ((float)NL - sScal[0]) + (-2.f * ld) - (float)NL);
    }
    if (tid < NL) m_all[((size_t)b * NT) * NL + tid] = sVecD[tid];
  }

  // ============================ scan t = 1..T-1 ===========================
  for (int t = 1; t < NT; ++t) {
    __syncthreads();
    for (int idx = tid; idx < NL * NR; idx += 256)
      sKt[idx >> 4][idx & 15] = Kg[(size_t)(b * NT + t) * NL * NR + idx];
    if (tid < NL) skt[tid] = kg[(size_t)(b * NT + t) * NL + tid];
    if (t + 1 < NT) {                        // prefetch next step's K tile
      const float* nk = &Kg[(size_t)(b * NT + t + 1) * NL * NR];
      if (tid < 64) __builtin_prefetch(nk + tid * 32, 0, 1);
      if (tid >= 64 && tid < 68)
        __builtin_prefetch(&kg[(size_t)(b * NT + t + 1) * NL] + (tid - 64) * 32, 0, 1);
    }
    __syncthreads();

    // --- GEMM1 (WMMA f16): hidden = tanh(z @ W1 + b1) -> GEMM2 A-frags ---
    for (int tile = wv; tile < 32; tile += 8) {
      int mt = tile >> 4, nt = tile & 15;
      v8f acc = {};
      for (int kt = 0; kt < 4; ++kt) {
        v16h a;
        int m = mt * 16 + ml;
        int kb = kt * 32 + ((lane >> 4) << 3);
        #pragma unroll
        for (int i = 0; i < 8; ++i) {
          a[i]     = (_Float16)sZ[m][kb + i];
          a[i + 8] = (_Float16)sZ[m][kb + 16 + i];
        }
        v16h bf = wsW1v[(kt * 16 + nt) * 32 + lane];
        acc = __builtin_amdgcn_wmma_f32_16x16x32_f16(false, a, false, bf,
                                                     (short)0, acc, false, false);
      }
      int row = (lane >> 4) * 8, n0 = nt * 16 + ml;
      float b1v = sB1[n0];
      int kt2 = n0 >> 5, kk = n0 & 31;
      int lane2, h2;
      if (kk < 8)       { lane2 = 0;  h2 = kk;      }
      else if (kk < 16) { lane2 = 16; h2 = kk - 8;  }
      else if (kk < 24) { lane2 = 0;  h2 = kk - 8;  }
      else              { lane2 = 16; h2 = kk - 16; }
      #pragma unroll
      for (int i = 0; i < 8; ++i) {
        int srow = row + i;
        float hval = tanhf(acc[i] + b1v);
        sH[(((mt * 8 + kt2) * 32) + lane2 + srow) * 16 + h2] = (_Float16)hval;
      }
    }
    __syncthreads();

    // --- GEMM2 (WMMA f16): m_theta = z + hidden @ W2 + b2 ---
    for (int tile = wv; tile < 16; tile += 8) {
      int mt = tile >> 3, nt2 = tile & 7;
      v8f acc = {};
      for (int kt2 = 0; kt2 < 8; ++kt2) {
        v16h a = *(const v16h*)&sH[((mt * 8 + kt2) * 32 + lane) * 16];
        v16h bf = wsW2v[(kt2 * 8 + nt2) * 32 + lane];
        acc = __builtin_amdgcn_wmma_f32_16x16x32_f16(false, a, false, bf,
                                                     (short)0, acc, false, false);
      }
      int row = (lane >> 4) * 8, l0 = nt2 * 16 + ml;
      float b2v = sB2[l0];
      #pragma unroll
      for (int i = 0; i < 8; ++i) {
        int s = mt * 16 + row + i;
        sMT[s][l0] = sZ[s][l0] + acc[i] + b2v;
      }
    }
    __syncthreads();

    // --- m_p, M_c ---
    if (tid < NL) {
      float mp = 0.f;
      for (int s = 0; s < NS; ++s) mp += sMT[s][tid];
      mp *= (1.0f / NS);
      sVecA[tid] = mp;
      for (int s = 0; s < NS; ++s) sMc[tid][s] = (sMT[s][tid] - mp) * sqrtSinv;
    }
    __syncthreads();

    // --- G = M_c^T Qinv M_c ; A32 = I + G   (WMMA f32 16x16x4) ---
    if (wv < 4) {
      int st = wv >> 1, tt = wv & 1;
      v8f acc = {};
      for (int k0 = 0; k0 < NL; k0 += 4) {
        v2f a, bf2;
        a[0]   = sQi[k0 + kh]     * sMc[k0 + kh][st * 16 + ml];
        a[1]   = sQi[k0 + kh + 1] * sMc[k0 + kh + 1][st * 16 + ml];
        bf2[0] = sMc[k0 + kh][tt * 16 + ml];
        bf2[1] = sMc[k0 + kh + 1][tt * 16 + ml];
        acc = __builtin_amdgcn_wmma_f32_16x16x4_f32(false, a, false, bf2,
                                                    (short)0, acc, false, false);
      }
      int row = (lane >> 4) * 8;
      #pragma unroll
      for (int i = 0; i < 8; ++i) {
        int s = st * 16 + row + i, t2 = tt * 16 + ml;
        sA32[s][t2] = acc[i] + ((s == t2) ? 1.0f : 0.0f);
      }
    }
    chol_lower(&sA32[0][0], NS, NS + 1, tid);        // Cp (lower) in place

    // --- h = Qinv*m_p - Qinv*(M_c (I+G)^{-1} M_c^T Qinv m_p) + k_t ---
    if (tid < NS) {
      float a_ = 0.f;
      for (int l = 0; l < NL; ++l) a_ += sMc[l][tid] * sQi[l] * sVecA[l];
      sSa[tid] = a_;
    }
    __syncthreads();
    if (wv == 0) {                                    // (CpCp^T)^{-1} via shfl
      float v = sSa[lane];
      for (int i2 = 0; i2 < NS; ++i2) {
        float vi = __shfl(v, i2) / sA32[i2][i2];
        if (lane == i2) v = vi;
        else if (lane > i2) v -= sA32[lane][i2] * vi;
      }
      for (int i2 = NS - 1; i2 >= 0; --i2) {
        float vi = __shfl(v, i2) / sA32[i2][i2];
        if (lane == i2) v = vi;
        else if (lane < i2) v -= sA32[i2][lane] * vi;
      }
      sSa[lane] = v;
    }
    __syncthreads();
    if (tid < NL) {
      float t4 = 0.f;
      for (int s = 0; s < NS; ++s) t4 += sMc[tid][s] * sSa[s];
      sVecB[tid] = sQi[tid] * (sVecA[tid] - t4) + skt[tid];
    }
    if (wv >= 2 && wv < 4) {                          // KtM = K^T M_c (WMMA)
      int tt = wv - 2;                                // N tile over s
      v8f acc = {};
      for (int k0 = 0; k0 < NL; k0 += 4) {
        v2f a, bf2;
        a[0]   = sKt[k0 + kh][ml];
        a[1]   = sKt[k0 + kh + 1][ml];
        bf2[0] = sMc[k0 + kh][tt * 16 + ml];
        bf2[1] = sMc[k0 + kh + 1][tt * 16 + ml];
        acc = __builtin_amdgcn_wmma_f32_16x16x4_f32(false, a, false, bf2,
                                                    (short)0, acc, false, false);
      }
      int row = (lane >> 4) * 8;
      #pragma unroll
      for (int i = 0; i < 8; ++i) sKtM[row + i][tt * 16 + ml] = acc[i];
    }
    __syncthreads();
    for (int idx = tid; idx < NR * NR; idx += 256) {  // I + KPK
      int r = idx >> 4, q = idx & 15;
      float a_ = (r == q) ? 1.0f : 0.0f;
      for (int s = 0; s < NS; ++s) a_ += sKtM[r][s] * sKtM[q][s];
      for (int l = 0; l < NL; ++l) a_ += sKt[l][r] * sQd[l] * sKt[l][q];
      sCf[r][q] = a_;
    }
    chol_lower(&sCf[0][0], NR, NR + 1, tid);          // Cf
    if (tid < NR) {                                   // inv = Cf^{-1}
      int c = tid; float x[NR];
      for (int i = 0; i < NR; ++i) {
        if (i < c) { x[i] = 0.f; continue; }
        float s_ = (i == c) ? 1.f : 0.f;
        for (int j = c; j < i; ++j) s_ -= sCf[i][j] * x[j];
        x[i] = s_ / sCf[i][i];
      }
      for (int i = 0; i < NR; ++i) sInv[i][c] = x[i];
    }
    __syncthreads();

    // --- u1h = bmv_Pp(M_c,Q,h); K_Psi ---
    if (tid < NS) {
      float a_ = 0.f;
      for (int l = 0; l < NL; ++l) a_ += sMc[l][tid] * sVecB[l];
      sSb[tid] = a_;
    }
    for (int idx = tid; idx < NL * NR; idx += 256) {
      int l = idx >> 4, r = idx & 15;
      float a_ = 0.f;
      for (int q = 0; q <= r; ++q) a_ += sKt[l][q] * sInv[r][q];
      sKPsi[l][r] = a_;
    }
    __syncthreads();
    if (tid < NL) {
      float a_ = 0.f;
      for (int s = 0; s < NS; ++s) a_ += sMc[tid][s] * sSb[s];
      sVecC[tid] = a_ + sQd[tid] * sVecB[tid];
    }
    __syncthreads();
    if (tid < NR) {                                   // a16 = K^T u1h
      float a_ = 0.f;
      for (int l = 0; l < NL; ++l) a_ += sKt[l][tid] * sVecC[l];
      sSa[tid] = a_;
    }
    __syncthreads();
    if (tid < NR) {
      float a_ = 0.f;
      for (int j = 0; j <= tid; ++j) a_ += sInv[tid][j] * sSa[j];
      sSb[tid] = a_;
    }
    __syncthreads();
    if (tid < NR) {
      float a_ = 0.f;
      for (int j = tid; j < NR; ++j) a_ += sInv[j][tid] * sSb[j];
      sSa[tid] = a_;
    }
    __syncthreads();
    if (tid < NL) {                                   // trip
      float a_ = 0.f;
      for (int r = 0; r < NR; ++r) a_ += sKt[tid][r] * sSa[r];
      sVecE[tid] = a_;
    }
    __syncthreads();
    if (tid < NS) {
      float a_ = 0.f;
      for (int l = 0; l < NL; ++l) a_ += sMc[l][tid] * sVecE[l];
      sSb[tid] = a_;
    }
    for (int idx = tid; idx < NS * NS; idx += 256)    // w_p1 draws
      sWp1[idx >> 5][idx & 31] = rngn(3u, t - 1, idx >> 5, b, idx & 31);
    for (int idx = tid; idx < NS * NR; idx += 256)    // w_f draws
      sWf[idx >> 4][idx & 15] = rngn(2u, t - 1, idx >> 4, b, idx & 15);
    __syncthreads();
    if (tid < NL) {                                   // m_f
      float a_ = 0.f;
      for (int s = 0; s < NS; ++s) a_ += sMc[tid][s] * sSb[s];
      sVecD[tid] = sVecC[tid] - (a_ + sQd[tid] * sVecE[tid]);
    }
    __syncthreads();

    // --- z_p_c = M_c @ w_p1 + Qsqrt*w_p2   (WMMA f32, M=s N=l K=t2) ---
    for (int tile = wv; tile < 16; tile += 8) {
      int st = tile >> 3, lt = tile & 7;
      v8f acc = {};
      for (int k0 = 0; k0 < NS; k0 += 4) {
        v2f a, bf2;
        a[0]   = sWp1[st * 16 + ml][k0 + kh];
        a[1]   = sWp1[st * 16 + ml][k0 + kh + 1];
        bf2[0] = sMc[lt * 16 + ml][k0 + kh];
        bf2[1] = sMc[lt * 16 + ml][k0 + kh + 1];
        acc = __builtin_amdgcn_wmma_f32_16x16x4_f32(false, a, false, bf2,
                                                    (short)0, acc, false, false);
      }
      int row = (lane >> 4) * 8, l = lt * 16 + ml;
      float qs = sQs[l];
      #pragma unroll
      for (int i = 0; i < 8; ++i) {
        int s = st * 16 + row + i;
        sZp[s][l] = acc[i] + qs * rngn(4u, t - 1, s, b, l);
      }
    }
    __syncthreads();

    // --- v1 = K^T z_p_c + w_f   (WMMA f32, M=s N=r K=l) ---
    if (wv < 2) {
      int st = wv;
      v8f acc = {};
      for (int k0 = 0; k0 < NL; k0 += 4) {
        v2f a, bf2;
        a[0]   = sZp[st * 16 + ml][k0 + kh];
        a[1]   = sZp[st * 16 + ml][k0 + kh + 1];
        bf2[0] = sKt[k0 + kh][ml];
        bf2[1] = sKt[k0 + kh + 1][ml];
        acc = __builtin_amdgcn_wmma_f32_16x16x4_f32(false, a, false, bf2,
                                                    (short)0, acc, false, false);
      }
      int row = (lane >> 4) * 8;
      #pragma unroll
      for (int i = 0; i < 8; ++i) {
        int s = st * 16 + row + i;
        sV1[s][ml] = acc[i] + sWf[s][ml];
      }
    }
    __syncthreads();
    for (int idx = tid; idx < NS * NR; idx += 256) {
      int s = idx >> 4, r = idx & 15;
      float a_ = 0.f;
      for (int j = 0; j <= r; ++j) a_ += sInv[r][j] * sV1[s][j];
      sY[s][r] = a_;
    }
    __syncthreads();
    for (int idx = tid; idx < NS * NR; idx += 256) {
      int s = idx >> 4, r = idx & 15;
      float a_ = 0.f;
      for (int j = r; j < NR; ++j) a_ += sInv[j][r] * sY[s][j];
      sX[s][r] = a_;
    }
    __syncthreads();

    // --- unscaled = (psi_solve v1) K^T   (WMMA f32, M=s N=l K=r) ---
    for (int tile = wv; tile < 16; tile += 8) {
      int st = tile >> 3, lt = tile & 7;
      v8f acc = {};
      for (int k0 = 0; k0 < NR; k0 += 4) {
        v2f a, bf2;
        a[0]   = sX[st * 16 + ml][k0 + kh];
        a[1]   = sX[st * 16 + ml][k0 + kh + 1];
        bf2[0] = sKt[lt * 16 + ml][k0 + kh];
        bf2[1] = sKt[lt * 16 + ml][k0 + kh + 1];
        acc = __builtin_amdgcn_wmma_f32_16x16x4_f32(false, a, false, bf2,
                                                    (short)0, acc, false, false);
      }
      int row = (lane >> 4) * 8, l = lt * 16 + ml;
      #pragma unroll
      for (int i = 0; i < 8; ++i) sUn[st * 16 + row + i][l] = acc[i];
    }
    __syncthreads();

    // --- TS = unscaled M_c   (WMMA f32, M=s N=t2 K=l) ---
    if (wv < 4) {
      int st = wv >> 1, tt = wv & 1;
      v8f acc = {};
      for (int k0 = 0; k0 < NL; k0 += 4) {
        v2f a, bf2;
        a[0]   = sUn[st * 16 + ml][k0 + kh];
        a[1]   = sUn[st * 16 + ml][k0 + kh + 1];
        bf2[0] = sMc[k0 + kh][tt * 16 + ml];
        bf2[1] = sMc[k0 + kh + 1][tt * 16 + ml];
        acc = __builtin_amdgcn_wmma_f32_16x16x4_f32(false, a, false, bf2,
                                                    (short)0, acc, false, false);
      }
      int row = (lane >> 4) * 8;
      #pragma unroll
      for (int i = 0; i < 8; ++i) sTS[st * 16 + row + i][tt * 16 + ml] = acc[i];
    }
    __syncthreads();

    // --- z_f = m_f + z_p_c - (M_c TS^T + Q*unscaled)  (WMMA f32) ---
    for (int tile = wv; tile < 16; tile += 8) {
      int st = tile >> 3, lt = tile & 7;
      v8f acc = {};
      for (int k0 = 0; k0 < NS; k0 += 4) {
        v2f a, bf2;
        a[0]   = sTS[st * 16 + ml][k0 + kh];
        a[1]   = sTS[st * 16 + ml][k0 + kh + 1];
        bf2[0] = sMc[lt * 16 + ml][k0 + kh];
        bf2[1] = sMc[lt * 16 + ml][k0 + kh + 1];
        acc = __builtin_amdgcn_wmma_f32_16x16x4_f32(false, a, false, bf2,
                                                    (short)0, acc, false, false);
      }
      int row = (lane >> 4) * 8, l = lt * 16 + ml;
      float md = sVecD[l], qd = sQd[l];
      #pragma unroll
      for (int i = 0; i < 8; ++i) {
        int s = st * 16 + row + i;
        float zf = md + sZp[s][l] - (acc[i] + qd * sUn[s][l]);
        sZ[s][l] = zf;
        z_all[(((size_t)s * NB + b) * NT + t) * NL + l] = zf;
      }
    }

    // --- KL terms ---
    float accT = 0.f;
    if (wv < 2) {                                     // tr1 (WMMA f32)
      int tt = wv;
      v8f acc = {};
      for (int k0 = 0; k0 < NL; k0 += 4) {
        v2f a, bf2;
        a[0]   = sKPsi[k0 + kh][ml];
        a[1]   = sKPsi[k0 + kh + 1][ml];
        bf2[0] = sMc[k0 + kh][tt * 16 + ml];
        bf2[1] = sMc[k0 + kh + 1][tt * 16 + ml];
        acc = __builtin_amdgcn_wmma_f32_16x16x4_f32(false, a, false, bf2,
                                                    (short)0, acc, false, false);
      }
      #pragma unroll
      for (int i = 0; i < 8; ++i) accT += acc[i] * acc[i];
    }
    for (int idx = tid; idx < NL * NR; idx += 256) {  // tr2
      int l = idx >> 4, r = idx & 15;
      float kp = sKPsi[l][r];
      accT += sQd[l] * kp * kp;
    }
    block_reduce_to(accT, 0);
    float accq = 0.f;
    if (tid < NL) {
      float dm = sVecD[tid] - sVecA[tid];
      accq = sQi[tid] * dm * dm;
    }
    block_reduce_to(accq, 1);
    if (tid < NS) {                                   // w = M_c^T Qinv dm
      float a_ = 0.f;
      for (int l = 0; l < NL; ++l)
        a_ += sQi[l] * sMc[l][tid] * (sVecD[l] - sVecA[l]);
      sSa[tid] = a_;
    }
    __syncthreads();
    if (wv == 0) {                                    // ||Cp^{-1} w||^2
      float v = sSa[lane];
      for (int i2 = 0; i2 < NS; ++i2) {
        float vi = __shfl(v, i2) / sA32[i2][i2];
        if (lane == i2) v = vi;
        else if (lane > i2) v -= sA32[lane][i2] * vi;
      }
      float sq = v * v;
      for (int off = 16; off > 0; off >>= 1) sq += __shfl_down(sq, off);
      if (lane == 0) sScal[2] = sq;
      if (lane == 0) {
        float ld = 0.f;
        for (int i = 0; i < NR; ++i) ld += logf(sInv[i][i] + EPSF);
        sScal[3] = -2.0f * ld;
      }
    }
    __syncthreads();
    if (tid == 0) {
      float qp = sScal[1] - sScal[2];
      kl_all[(size_t)b * NT + t] =
          0.5f * (qp + ((float)NL - sScal[0]) + sScal[3] - (float)NL);
    }
    if (tid < NL) m_all[((size_t)b * NT + t) * NL + tid] = sVecD[tid];
  }
}

extern "C" void kernel_launch(void* const* d_in, const int* in_sizes, int n_in,
                              void* d_out, int out_size, void* d_ws, size_t ws_size,
                              hipStream_t stream) {
  (void)in_sizes; (void)n_in; (void)out_size; (void)ws_size;
  const float* k       = (const float*)d_in[0];
  const float* K       = (const float*)d_in[1];
  const float* log_Q   = (const float*)d_in[2];
  const float* m_0     = (const float*)d_in[3];
  const float* log_Q_0 = (const float*)d_in[4];
  const float* W1      = (const float*)d_in[5];
  const float* b1      = (const float*)d_in[6];
  const float* W2      = (const float*)d_in[7];
  const float* b2      = (const float*)d_in[8];
  _Float16* wsW = (_Float16*)d_ws;   // 128 KB: swizzled f16 W1 then W2

  nlf_swizzle<<<NB, 256, 0, stream>>>(W1, W2, wsW);
  nlf_filter<<<NB, 256, 0, stream>>>(k, K, log_Q, m_0, log_Q_0, b1, b2, wsW,
                                     (float*)d_out);
}
